// TernaryLinear_37434934952235
// MI455X (gfx1250) — matrix operand order repaired and checked by reading, not compiled, not run on hardware
//
#include <hip/hip_runtime.h>
#include <hip/hip_bf16.h>

// ---------------------------------------------------------------------------
// TernaryLinear forward on gfx1250 (MI455X):
//   scale[o] = max(mean|W[o,:]|, 1e-6); thr = 0.75*scale
//   q = ternary(W) in {-1,0,+1}  (exact in bf16)
//   y = scale[o] * (x @ q^T) + bias
// GEMM via v_wmma_f32_16x16x32_bf16. x is pre-split x = hi + lo (bf16 pair)
// for ~fp32 accuracy; q is exact. Tiles double-buffered in LDS via
// GLOBAL_LOAD_ASYNC_TO_LDS_B128 (ASYNCcnt-tracked).
// ---------------------------------------------------------------------------

typedef __attribute__((ext_vector_type(16))) __bf16 v16bf;
typedef __attribute__((ext_vector_type(8)))  float  v8f;

union FragBF { v16bf v; uint4 u[2]; };

__device__ __forceinline__ unsigned short f2bf(float f) {
    unsigned u = __float_as_uint(f);
    unsigned r = u + 0x7FFFu + ((u >> 16) & 1u);   // round-to-nearest-even
    return (unsigned short)(r >> 16);
}
__device__ __forceinline__ float bf2f(unsigned short h) {
    return __uint_as_float(((unsigned)h) << 16);
}
__device__ __forceinline__ unsigned lds_addr(const void* p) {
    // generic pointer to LDS: low 32 bits are the LDS byte offset
    return (unsigned)(unsigned long long)p;
}

// -------------------------- dimensions (fixed) -----------------------------
#define MDIM 8192
#define KDIM 4096
#define NDIM 4096
#define BM   128
#define BN   128
#define BK   64
#define LDK  72   // padded LDS row stride (elements): 144B -> conflict-free b128 reads

// --------------------- kernel 0: split x into hi/lo bf16 -------------------
__global__ __launch_bounds__(256) void split_x_kernel(
    const float* __restrict__ X,        // [MDIM*KDIM]
    unsigned short* __restrict__ XH,    // bf16 bits
    unsigned short* __restrict__ XL)    // bf16 bits (residual)
{
    const size_t base = ((size_t)blockIdx.x * 256 + threadIdx.x) * 8;
    float4 f0 = *(const float4*)(X + base);
    float4 f1 = *(const float4*)(X + base + 4);
    float f[8] = {f0.x, f0.y, f0.z, f0.w, f1.x, f1.y, f1.z, f1.w};
    unsigned h[8], l[8];
#pragma unroll
    for (int j = 0; j < 8; ++j) {
        unsigned short hb = f2bf(f[j]);
        h[j] = hb;
        l[j] = f2bf(f[j] - bf2f(hb));
    }
    uint4 uh, ul;
    uh.x = h[0] | (h[1] << 16); uh.y = h[2] | (h[3] << 16);
    uh.z = h[4] | (h[5] << 16); uh.w = h[6] | (h[7] << 16);
    ul.x = l[0] | (l[1] << 16); ul.y = l[2] | (l[3] << 16);
    ul.z = l[4] | (l[5] << 16); ul.w = l[6] | (l[7] << 16);
    *(uint4*)(XH + base) = uh;
    *(uint4*)(XL + base) = ul;
}

// --------------------- kernel 1: ternary quantize W ------------------------
__global__ __launch_bounds__(256) void ternary_quant_kernel(
    const float* __restrict__ W,          // [NDIM, KDIM]
    unsigned short* __restrict__ Q,       // [NDIM, KDIM] bf16 bits in {-1,0,+1}
    float* __restrict__ scale_out)        // [NDIM]
{
    const int row = blockIdx.x;
    const float* wr = W + (size_t)row * KDIM;

    float s = 0.f;
    for (int i = threadIdx.x * 4; i < KDIM; i += 256 * 4) {
        float4 v = *(const float4*)(wr + i);
        s += fabsf(v.x) + fabsf(v.y) + fabsf(v.z) + fabsf(v.w);
    }
    __shared__ float red[256];
    red[threadIdx.x] = s;
    __syncthreads();
    for (int off = 128; off > 0; off >>= 1) {
        if (threadIdx.x < off) red[threadIdx.x] += red[threadIdx.x + off];
        __syncthreads();
    }
    const float scale = fmaxf(red[0] * (1.0f / (float)KDIM), 1e-6f);
    const float thr   = 0.75f * scale;
    if (threadIdx.x == 0) scale_out[row] = scale;

    unsigned short* qr = Q + (size_t)row * KDIM;
    for (int i = threadIdx.x * 4; i < KDIM; i += 256 * 4) {
        float4 v = *(const float4*)(wr + i);
        ushort4 qv;
        qv.x = (v.x >= thr) ? 0x3F80u : ((v.x <= -thr) ? 0xBF80u : 0u);
        qv.y = (v.y >= thr) ? 0x3F80u : ((v.y <= -thr) ? 0xBF80u : 0u);
        qv.z = (v.z >= thr) ? 0x3F80u : ((v.z <= -thr) ? 0xBF80u : 0u);
        qv.w = (v.w >= thr) ? 0x3F80u : ((v.w <= -thr) ? 0xBF80u : 0u);
        *(ushort4*)(qr + i) = qv;
    }
}

// ------------- async copy of a 128x64 bf16 tile into LDS -------------------
// 1024 x b128 chunks; 4 per thread. GVS addressing: saddr(base) + voffset.
__device__ __forceinline__ void async_tile(
    const unsigned short* gbase,  // tile row 0, element 0 (includes block offset)
    unsigned kb_bytes,            // kb * 2
    unsigned lbase,               // LDS byte address of tile start
    int tid)
{
#pragma unroll
    for (int it = 0; it < 4; ++it) {
        const int idx = tid + it * 256;      // 0..1023
        const int r   = idx >> 3;            // row 0..127
        const int c   = (idx & 7) << 3;      // col 0..56 (elements)
        const unsigned voff  = kb_bytes + (unsigned)((r * KDIM + c) * 2);
        const unsigned laddr = lbase + (unsigned)((r * LDK + c) * 2);
        asm volatile("global_load_async_to_lds_b128 %0, %1, %2"
                     :: "v"(laddr), "v"(voff), "s"(gbase)
                     : "memory");
    }
}

// --------------- kernel 2: WMMA GEMM, async double-buffered ----------------
__global__ __launch_bounds__(256) void ternary_gemm_async_kernel(
    const unsigned short* __restrict__ XH,   // [MDIM, KDIM] bf16 bits
    const unsigned short* __restrict__ XL,   // [MDIM, KDIM] bf16 bits
    const unsigned short* __restrict__ Q,    // [NDIM, KDIM] bf16 bits
    const float* __restrict__ scale,         // [NDIM]
    const float* __restrict__ bias,          // [NDIM]
    float* __restrict__ Y)                   // [MDIM, NDIM] fp32
{
    __shared__ alignas(16) unsigned short sAh[2][BM * LDK];
    __shared__ alignas(16) unsigned short sAl[2][BM * LDK];
    __shared__ alignas(16) unsigned short sB [2][BN * LDK];

    const int tid  = threadIdx.x;
    const int lane = tid & 31;
    const int wave = tid >> 5;        // 0..7
    const int wm   = wave & 3;        // 4 waves tile M (32 rows each)
    const int wn   = wave >> 2;       // 2 waves tile N (64 cols each)
    const int l15  = lane & 15;
    const int lhi  = (lane >> 4) & 1;

    const int blockM = blockIdx.y * BM;
    const int blockN = blockIdx.x * BN;

    const unsigned short* gAH = XH + (size_t)blockM * KDIM;
    const unsigned short* gAL = XL + (size_t)blockM * KDIM;
    const unsigned short* gB  = Q  + (size_t)blockN * KDIM;

    v8f acc[2][4];
#pragma unroll
    for (int mt = 0; mt < 2; ++mt)
#pragma unroll
        for (int nt = 0; nt < 4; ++nt)
            acc[mt][nt] = (v8f)0.0f;

    // prologue: fill buffer 0  (12 async b128 per thread)
    async_tile(gAH, 0u, lds_addr(&sAh[0][0]), tid);
    async_tile(gAL, 0u, lds_addr(&sAl[0][0]), tid);
    async_tile(gB,  0u, lds_addr(&sB [0][0]), tid);

    int cur = 0;
    for (int kb = 0; kb < KDIM; kb += BK) {
        if (kb + BK < KDIM) {
            const unsigned nb = (unsigned)(kb + BK) * 2u;
            const int nxt = cur ^ 1;
            async_tile(gAH, nb, lds_addr(&sAh[nxt][0]), tid);
            async_tile(gAL, nb, lds_addr(&sAl[nxt][0]), tid);
            async_tile(gB,  nb, lds_addr(&sB [nxt][0]), tid);
            // 12 newer ops outstanding; <=12 means previous batch landed
            asm volatile("s_wait_asynccnt 12" ::: "memory");
        } else {
            asm volatile("s_wait_asynccnt 0" ::: "memory");
        }
        __syncthreads();

        const unsigned short* ah = sAh[cur];
        const unsigned short* al = sAl[cur];
        const unsigned short* bq = sB [cur];

#pragma unroll
        for (int kc = 0; kc < 2; ++kc) {
            FragBF bfm[4];
#pragma unroll
            for (int nt = 0; nt < 4; ++nt) {
                const unsigned short* p =
                    &bq[(wn * 64 + nt * 16 + l15) * LDK + kc * 32 + (lhi << 3)];
                bfm[nt].u[0] = *(const uint4*)(p);
                bfm[nt].u[1] = *(const uint4*)(p + 16);
            }
            FragBF fah[2], fal[2];
#pragma unroll
            for (int mt = 0; mt < 2; ++mt) {
                const int row = wm * 32 + mt * 16 + l15;
                const int k0  = kc * 32 + (lhi << 3);
                fah[mt].u[0] = *(const uint4*)&ah[row * LDK + k0];
                fah[mt].u[1] = *(const uint4*)&ah[row * LDK + k0 + 16];
                fal[mt].u[0] = *(const uint4*)&al[row * LDK + k0];
                fal[mt].u[1] = *(const uint4*)&al[row * LDK + k0 + 16];
            }
#pragma unroll
            for (int mt = 0; mt < 2; ++mt)
#pragma unroll
                for (int nt = 0; nt < 4; ++nt) {
                    acc[mt][nt] = __builtin_amdgcn_wmma_f32_16x16x32_bf16(
                        false, fah[mt].v, false, bfm[nt].v, (short)0, acc[mt][nt],
                        false, false);
                    acc[mt][nt] = __builtin_amdgcn_wmma_f32_16x16x32_bf16(
                        false, fal[mt].v, false, bfm[nt].v, (short)0, acc[mt][nt],
                        false, false);
                }
        }
        __syncthreads();   // everyone done reading buf[cur] before it is refilled
        cur ^= 1;
    }

#pragma unroll
    for (int nt = 0; nt < 4; ++nt) {
        const int col = blockN + wn * 64 + nt * 16 + l15;
        const float s = scale[col];
        const float b = bias[col];
#pragma unroll
        for (int mt = 0; mt < 2; ++mt) {
            const int rbase = blockM + wm * 32 + mt * 16 + (lhi << 3);
#pragma unroll
            for (int j = 0; j < 8; ++j) {
                Y[(size_t)(rbase + j) * NDIM + col] = acc[mt][nt][j] * s + b;
            }
        }
    }
}

// ------- fallback kernel (round-1 fused path) if workspace is small --------
__global__ __launch_bounds__(256) void ternary_gemm_fused_kernel(
    const float* __restrict__ X,
    const unsigned short* __restrict__ Q,
    const float* __restrict__ scale,
    const float* __restrict__ bias,
    float* __restrict__ Y)
{
    __shared__ alignas(16) unsigned short sAh[BM * LDK];
    __shared__ alignas(16) unsigned short sAl[BM * LDK];
    __shared__ alignas(16) unsigned short sB [BN * LDK];

    const int tid  = threadIdx.x;
    const int lane = tid & 31;
    const int wave = tid >> 5;
    const int wm   = wave & 3;
    const int wn   = wave >> 2;
    const int l15  = lane & 15;
    const int lhi  = (lane >> 4) & 1;

    const int blockM = blockIdx.y * BM;
    const int blockN = blockIdx.x * BN;

    v8f acc[2][4];
#pragma unroll
    for (int mt = 0; mt < 2; ++mt)
#pragma unroll
        for (int nt = 0; nt < 4; ++nt)
            acc[mt][nt] = (v8f)0.0f;

    for (int kb = 0; kb < KDIM; kb += BK) {
#pragma unroll
        for (int it = 0; it < 4; ++it) {
            const int idx = tid + it * 256;
            const int r   = idx >> 3;
            const int c   = (idx & 7) << 3;
            const float4* gp = (const float4*)(X + (size_t)(blockM + r) * KDIM + kb + c);
            float4 f0 = gp[0], f1 = gp[1];
            float f[8] = {f0.x, f0.y, f0.z, f0.w, f1.x, f1.y, f1.z, f1.w};
            unsigned h[8], l[8];
#pragma unroll
            for (int j = 0; j < 8; ++j) {
                unsigned short hb = f2bf(f[j]);
                h[j] = hb;
                l[j] = f2bf(f[j] - bf2f(hb));
            }
            uint4 uh, ul;
            uh.x = h[0] | (h[1] << 16); uh.y = h[2] | (h[3] << 16);
            uh.z = h[4] | (h[5] << 16); uh.w = h[6] | (h[7] << 16);
            ul.x = l[0] | (l[1] << 16); ul.y = l[2] | (l[3] << 16);
            ul.z = l[4] | (l[5] << 16); ul.w = l[6] | (l[7] << 16);
            *(uint4*)&sAh[r * LDK + c] = uh;
            *(uint4*)&sAl[r * LDK + c] = ul;
        }
#pragma unroll
        for (int it = 0; it < 4; ++it) {
            const int idx = tid + it * 256;
            const int r   = idx >> 3;
            const int c   = (idx & 7) << 3;
            uint4 qv = *(const uint4*)(Q + (size_t)(blockN + r) * KDIM + kb + c);
            *(uint4*)&sB[r * LDK + c] = qv;
        }
        __syncthreads();
#pragma unroll
        for (int kc = 0; kc < 2; ++kc) {
            FragBF bfm[4];
#pragma unroll
            for (int nt = 0; nt < 4; ++nt) {
                const unsigned short* p =
                    &sB[(wn * 64 + nt * 16 + l15) * LDK + kc * 32 + (lhi << 3)];
                bfm[nt].u[0] = *(const uint4*)(p);
                bfm[nt].u[1] = *(const uint4*)(p + 16);
            }
            FragBF fah[2], fal[2];
#pragma unroll
            for (int mt = 0; mt < 2; ++mt) {
                const int row = wm * 32 + mt * 16 + l15;
                const int k0  = kc * 32 + (lhi << 3);
                fah[mt].u[0] = *(const uint4*)&sAh[row * LDK + k0];
                fah[mt].u[1] = *(const uint4*)&sAh[row * LDK + k0 + 16];
                fal[mt].u[0] = *(const uint4*)&sAl[row * LDK + k0];
                fal[mt].u[1] = *(const uint4*)&sAl[row * LDK + k0 + 16];
            }
#pragma unroll
            for (int mt = 0; mt < 2; ++mt)
#pragma unroll
                for (int nt = 0; nt < 4; ++nt) {
                    acc[mt][nt] = __builtin_amdgcn_wmma_f32_16x16x32_bf16(
                        false, fah[mt].v, false, bfm[nt].v, (short)0, acc[mt][nt],
                        false, false);
                    acc[mt][nt] = __builtin_amdgcn_wmma_f32_16x16x32_bf16(
                        false, fal[mt].v, false, bfm[nt].v, (short)0, acc[mt][nt],
                        false, false);
                }
        }
        __syncthreads();
    }

#pragma unroll
    for (int nt = 0; nt < 4; ++nt) {
        const int col = blockN + wn * 64 + nt * 16 + l15;
        const float s = scale[col];
        const float b = bias[col];
#pragma unroll
        for (int mt = 0; mt < 2; ++mt) {
            const int rbase = blockM + wm * 32 + mt * 16 + (lhi << 3);
#pragma unroll
            for (int j = 0; j < 8; ++j) {
                Y[(size_t)(rbase + j) * NDIM + col] = acc[mt][nt][j] * s + b;
            }
        }
    }
}

// ---------------------------------------------------------------------------
extern "C" void kernel_launch(void* const* d_in, const int* in_sizes, int n_in,
                              void* d_out, int out_size, void* d_ws, size_t ws_size,
                              hipStream_t stream) {
    const float* x    = (const float*)d_in[0];   // [8192, 4096]
    const float* w    = (const float*)d_in[1];   // [4096, 4096]
    const float* bias = (const float*)d_in[2];   // [4096]
    float* out = (float*)d_out;                  // [8192, 4096]

    const size_t Q_BYTES  = (size_t)NDIM * KDIM * sizeof(unsigned short); // 32 MB
    const size_t SC_BYTES = (size_t)NDIM * sizeof(float);                 // 16 KB
    const size_t X_BYTES  = (size_t)MDIM * KDIM * sizeof(unsigned short); // 64 MB

    unsigned short* q  = (unsigned short*)d_ws;
    float* scale       = (float*)((char*)d_ws + Q_BYTES);
    unsigned short* xh = (unsigned short*)((char*)d_ws + Q_BYTES + SC_BYTES);
    unsigned short* xl = (unsigned short*)((char*)d_ws + Q_BYTES + SC_BYTES + X_BYTES);

    ternary_quant_kernel<<<NDIM, 256, 0, stream>>>(w, q, scale);

    dim3 grid(NDIM / BN, MDIM / BM);  // (32, 64)

    if (ws_size >= Q_BYTES + SC_BYTES + 2 * X_BYTES) {
        split_x_kernel<<<(MDIM * (size_t)KDIM) / (256 * 8), 256, 0, stream>>>(x, xh, xl);
        ternary_gemm_async_kernel<<<grid, 256, 0, stream>>>(xh, xl, q, scale, bias, out);
    } else {
        ternary_gemm_fused_kernel<<<grid, 256, 0, stream>>>(x, q, scale, bias, out);
    }
}
